// CRF_6330781794343
// MI455X (gfx1250) — compile-verified, matrix operand order checked by reading, not run on hardware
//
#include <hip/hip_runtime.h>

typedef __attribute__((ext_vector_type(16))) _Float16 v16h;
typedef __attribute__((ext_vector_type(8)))  float    v8f;
typedef __attribute__((ext_vector_type(4)))  float    v4f;

constexpr int kB  = 16384;
constexpr int kT  = 14;
constexpr int kE  = 128;
constexpr int kL  = 26;
constexpr int kLP = 32;               // labels padded to 2 WMMA col tiles
constexpr int kNB = 16;               // batch items per block
constexpr int kROWS = kNB * kT;       // 224 GEMM rows per block
constexpr int kRT   = kROWS / 16;     // 14 row tiles per block

// ---------------------------------------------------------------------------
// Prep: fold conv (5x5 SAME on 16x8, cross-correlation) into the emission
// matrix:  M[l][e_in] = sum_{e_out in nbhd} Wmat[l][e_out] * k[...]
// and biasL[l] = conv_b * sum_e Wmat[l][e].  M stored f16 (32x128, padded).
// ---------------------------------------------------------------------------
__global__ void crf_prep(const float* __restrict__ conv_w,
                         const float* __restrict__ conv_b,
                         const float* __restrict__ params,
                         _Float16* __restrict__ Mh,
                         float* __restrict__ biasL)
{
    int tid = blockIdx.x * blockDim.x + threadIdx.x;   // 0..4095
    if (tid < kLP * kE) {
        int l = tid >> 7;
        int e = tid & 127;
        float m = 0.0f;
        if (l < kL) {
            int hi = e >> 3, wi = e & 7;               // input pixel (h,w), W=8
            for (int ho = hi - 2; ho <= hi + 2; ++ho) {
                if (ho < 0 || ho >= 16) continue;
                for (int wo = wi - 2; wo <= wi + 2; ++wo) {
                    if (wo < 0 || wo >= 8) continue;
                    m += params[l * kE + ho * 8 + wo] *
                         conv_w[(hi - ho + 2) * 5 + (wi - wo + 2)];
                }
            }
        }
        Mh[tid] = (_Float16)m;
    }
    if (tid < kLP) {
        float s = 0.0f;
        if (tid < kL) {
            for (int e = 0; e < kE; ++e) s += params[tid * kE + e];
            s *= conv_b[0];
        }
        biasL[tid] = s;
    }
}

// ---------------------------------------------------------------------------
// Main fused kernel: WMMA emission GEMM -> LDS -> Viterbi -> backtrack.
// 128 threads (4 wave32) per block, 16 batch items per block.
// ---------------------------------------------------------------------------
__global__ __launch_bounds__(128) void crf_main(
    const float*    __restrict__ X,        // (B*T, 128) f32
    const float*    __restrict__ params,   // Trans at offset L*E
    const _Float16* __restrict__ Mh,       // (32, 128) f16
    const float*    __restrict__ biasL,    // (32,) f32
    int*            __restrict__ preds_out,// (B, T) int32
    float*          __restrict__ score_out)// (B,)  f32
{
    __shared__ float         s_emis[kROWS][kLP];        // 28672 B
    __shared__ float         s_trans[kL][kLP];          //  3328 B (padded rows)
    __shared__ float         s_alpha[2][kNB][kLP];      //  4096 B
    __shared__ unsigned char s_bp[kNB][kT - 1][kLP];    //  6656 B

    const int tid  = threadIdx.x;
    const int wave = tid >> 5;
    const int lane = tid & 31;
    const int lo16 = lane & 15;
    const int hiH  = (lane >> 4) & 1;     // 0: lanes 0-15, 1: lanes 16-31

    // --- transition matrix -> LDS (row stride 32 for bank-conflict-free j) --
    for (int idx = tid; idx < kL * kL; idx += 128)
        s_trans[idx / kL][idx % kL] = params[kL * kE + idx];

    // --- B fragments (emission matrix), both column tiles, kept in VGPRs ----
    // B is KxN (32x16) per tile; lane half selects K range per ISA layout:
    //   lanes 0-15 : halves 0-7 -> K 0-7,  halves 8-15 -> K 16-23
    //   lanes 16-31: halves 0-7 -> K 8-15, halves 8-15 -> K 24-31
    v16h bfrag[2][4];
    for (int c = 0; c < 2; ++c) {
        const int n = c * 16 + lo16;                 // label (column) index
        for (int kk = 0; kk < 4; ++kk) {
            v16h f;
#pragma unroll
            for (int h = 0; h < 16; ++h) {
                int kl = (h < 8) ? h : (h + 8);
                int k  = kk * 32 + kl + hiH * 8;
                f[h] = Mh[n * kE + k];
            }
            bfrag[c][kk] = f;
        }
    }
    const float bias0 = biasL[lo16];
    const float bias1 = biasL[16 + lo16];

    const long rowBase = (long)blockIdx.x * kROWS;   // first (b,t) row

    // --- emission GEMM: each wave does row tiles wave, wave+4, ... ----------
    for (int rt = wave; rt < kRT; rt += 4) {
        const int rows0 = rt * 16;
        const float* rp = X + (rowBase + rows0 + lo16) * (long)kE;
        __builtin_prefetch(rp + 64 * kE, 0, 1);      // next tile for this wave

        v16h afrag[4];
#pragma unroll
        for (int kk = 0; kk < 4; ++kk) {
            const int kb = kk * 32 + hiH * 8;
            v4f a0 = *(const v4f*)(rp + kb);
            v4f a1 = *(const v4f*)(rp + kb + 4);
            v4f a2 = *(const v4f*)(rp + kb + 16);
            v4f a3 = *(const v4f*)(rp + kb + 20);
            v16h f;
#pragma unroll
            for (int i2 = 0; i2 < 4; ++i2) {
                f[i2]      = (_Float16)a0[i2];
                f[4 + i2]  = (_Float16)a1[i2];
                f[8 + i2]  = (_Float16)a2[i2];
                f[12 + i2] = (_Float16)a3[i2];
            }
            afrag[kk] = f;
        }

        v8f acc0 = {};
        v8f acc1 = {};
#pragma unroll
        for (int kk = 0; kk < 4; ++kk) {
            acc0 = __builtin_amdgcn_wmma_f32_16x16x32_f16(
                false, afrag[kk], false, bfrag[0][kk],
                (short)0, acc0, false, false);
            acc1 = __builtin_amdgcn_wmma_f32_16x16x32_f16(
                false, afrag[kk], false, bfrag[1][kk],
                (short)0, acc1, false, false);
        }

        // C layout: VGPR r -> (M = r + 8*hiH, N = lo16) per ISA table
#pragma unroll
        for (int r = 0; r < 8; ++r) {
            const int row = rows0 + r + hiH * 8;
            s_emis[row][lo16]      = acc0[r] + bias0;
            s_emis[row][16 + lo16] = acc1[r] + bias1;
        }
    }
    __syncthreads();

    // --- Viterbi forward: lane = label j, each wave owns 4 batch items ------
    const int j = lane;
    for (int b = wave; b < kNB; b += 4)
        s_alpha[0][b][j] = (j < kL) ? s_emis[b * kT + 0][j] : -1e30f;
    __syncthreads();

    int cur = 0;
    for (int t = 1; t < kT; ++t) {
        for (int b = wave; b < kNB; b += 4) {
            float best = -1e30f;
            int   arg  = 0;
            if (j < kL) {
                for (int i = 0; i < kL; ++i) {
                    float s = s_alpha[cur][b][i] + s_trans[i][j];
                    if (s > best) { best = s; arg = i; }
                }
                best += s_emis[b * kT + t][j];
            }
            s_bp[b][t - 1][j]       = (unsigned char)arg;
            s_alpha[cur ^ 1][b][j]  = best;
        }
        __syncthreads();
        cur ^= 1;
    }

    // --- backtrack: one thread per batch item -------------------------------
    if (tid < kNB) {
        const int b = tid;
        float best = s_alpha[cur][b][0];
        int   lbl  = 0;
        for (int i = 1; i < kL; ++i) {
            float v = s_alpha[cur][b][i];
            if (v > best) { best = v; lbl = i; }
        }
        const long gb = (long)blockIdx.x * kNB + b;
        score_out[gb] = best;
        preds_out[gb * kT + (kT - 1)] = lbl;
        for (int t = kT - 1; t >= 1; --t) {
            lbl = s_bp[b][t - 1][lbl];
            preds_out[gb * kT + (t - 1)] = lbl;
        }
    }
}

// ---------------------------------------------------------------------------
extern "C" void kernel_launch(void* const* d_in, const int* in_sizes, int n_in,
                              void* d_out, int out_size, void* d_ws, size_t ws_size,
                              hipStream_t stream)
{
    const float* X      = (const float*)d_in[0];   // (B,T,E) f32
    const float* conv_w = (const float*)d_in[1];   // (1,1,5,5)
    const float* conv_b = (const float*)d_in[2];   // (1,)
    const float* params = (const float*)d_in[3];   // (L*E + L*L,)

    _Float16* Mh    = (_Float16*)d_ws;                                  // 8192 B
    float*    biasL = (float*)((char*)d_ws + kLP * kE * sizeof(_Float16));

    crf_prep<<<16, 256, 0, stream>>>(conv_w, conv_b, params, Mh, biasL);

    int*   preds = (int*)d_out;
    float* score = (float*)d_out + (size_t)kB * kT;

    crf_main<<<kB / kNB, 128, 0, stream>>>(X, params, Mh, biasL, preds, score);
}
